// MultiHeadSelfAttention_21663815041799
// MI455X (gfx1250) — matrix operand order repaired and checked by reading, not compiled, not run on hardware
//
#include <hip/hip_runtime.h>

typedef __bf16 bf16_t;
typedef __attribute__((ext_vector_type(16))) __bf16 v16bf;
typedef __attribute__((ext_vector_type(8)))  __bf16 v8bf;
typedef __attribute__((ext_vector_type(8)))  float  v8f;

#define DM   1024
#define NH   16
#define DK   64
#define SQ   2048
#define NB   4
#define NTOK (NB * SQ)   // 8192 tokens

#define LDS_STRIDE 72          // 144 B row pitch: 16B-aligned, conflict-free
#define A_BUF (128 * LDS_STRIDE)
#define B_BUF (64 * LDS_STRIDE)

__device__ __forceinline__ bf16_t f2bf(float f) {
  unsigned u = __builtin_bit_cast(unsigned, f);
  unsigned r = u + 0x7FFFu + ((u >> 16) & 1u);   // round to nearest even
  unsigned short h = (unsigned short)(r >> 16);
  return __builtin_bit_cast(bf16_t, h);
}

union AF { v16bf v; v8bf h[2]; };

// A-frag (16x32 bf16): lane holds row M=lane%16; K = {8h..8h+7} U {16+8h..16+8h+7}
__device__ __forceinline__ v16bf ldA(const bf16_t* base, int stride, int lane) {
  int m = lane & 15, hf = lane >> 4;
  AF f;
  f.h[0] = *(const v8bf*)(base + m * stride + 8 * hf);
  f.h[1] = *(const v8bf*)(base + m * stride + 16 + 8 * hf);
  return f.v;
}
// B-frag (32x16 bf16) from K-contiguous (transposed) storage: lane holds col
// N=lane%16, K = {8h..8h+7} U {16+8h..16+8h+7}.
__device__ __forceinline__ v16bf ldB(const bf16_t* base, int stride, int lane) {
  int n = lane & 15, hf = lane >> 4;
  const bf16_t* p = base + n * stride + 8 * hf;
  AF f;
  f.h[0] = *(const v8bf*)(p);
  f.h[1] = *(const v8bf*)(p + 16);
  return f.v;
}

#define WMMA_BF16(a, b, c) \
  __builtin_amdgcn_wmma_f32_16x16x32_bf16(false, (a), false, (b), (short)0, (c), false, false)

// Async copy of one 16-byte chunk: global -> LDS, no VGPR data, ASYNCcnt-tracked.
__device__ __forceinline__ void async_b128(const bf16_t* g, const bf16_t* l) {
  unsigned loff = (unsigned)(size_t)l;            // low 32 bits of flat LDS addr = LDS offset
  unsigned long long ga = (unsigned long long)(size_t)g;
  asm volatile("global_load_async_to_lds_b128 %0, %1, off" :: "v"(loff), "v"(ga) : "memory");
}
__device__ __forceinline__ void wait_async(int n) {
  if (n == 0) asm volatile("s_wait_asynccnt 0x0" ::: "memory");
  else        asm volatile("s_wait_asynccnt 0x6" ::: "memory");
}

// Issue one 128x64 A tile + 64x64 B tile (bf16) into LDS buffer `buf`.
// 6 async instructions per wave (4 for A, 2 for B).
__device__ __forceinline__ void issue_tile(const bf16_t* Ag, const bf16_t* Bg,
                                           bf16_t* As, bf16_t* Bs, int buf,
                                           int m0, int n0, int k0) {
  const int tid = threadIdx.x;
#pragma unroll
  for (int c = 0; c < 4; ++c) {          // A: 1024 chunks of 8 elems
    int id = tid + 256 * c;
    int row = id >> 3, col = (id & 7) * 8;
    async_b128(Ag + (size_t)(m0 + row) * DM + k0 + col,
               As + buf * A_BUF + row * LDS_STRIDE + col);
  }
#pragma unroll
  for (int c = 0; c < 2; ++c) {          // B: 512 chunks of 8 elems
    int id = tid + 256 * c;
    int row = id >> 3, col = (id & 7) * 8;
    async_b128(Bg + (size_t)(n0 + row) * DM + k0 + col,
               Bs + buf * B_BUF + row * LDS_STRIDE + col);
  }
}

// Double-buffered BK=64 GEMM core: acc += A[128x1024] @ B^T for this block tile.
__device__ __forceinline__ void gemm_core(const bf16_t* Ag, const bf16_t* Bg,
                                          bf16_t* As, bf16_t* Bs,
                                          int m0, int n0, int lane, int wm, int wn,
                                          v8f acc[2][2]) {
  issue_tile(Ag, Bg, As, Bs, 0, m0, n0, 0);
  issue_tile(Ag, Bg, As, Bs, 1, m0, n0, 64);
  const int NT = DM / 64;   // 16 k-tiles
  for (int i = 0; i < NT; ++i) {
    wait_async((i < NT - 1) ? 6 : 0);    // oldest tile's 6 asyncs complete
    __syncthreads();                     // everyone's portion of tile i landed
    const bf16_t* Ab = As + (i & 1) * A_BUF;
    const bf16_t* Bb = Bs + (i & 1) * B_BUF;
#pragma unroll
    for (int j = 0; j < 2; ++j) {        // two 32-deep k chunks
      v16bf a0 = ldA(Ab + (wm * 32) * LDS_STRIDE + 32 * j, LDS_STRIDE, lane);
      v16bf a1 = ldA(Ab + (wm * 32 + 16) * LDS_STRIDE + 32 * j, LDS_STRIDE, lane);
      v16bf b0 = ldB(Bb + (wn * 32) * LDS_STRIDE + 32 * j, LDS_STRIDE, lane);
      v16bf b1 = ldB(Bb + (wn * 32 + 16) * LDS_STRIDE + 32 * j, LDS_STRIDE, lane);
      acc[0][0] = WMMA_BF16(a0, b0, acc[0][0]);
      acc[0][1] = WMMA_BF16(a0, b1, acc[0][1]);
      acc[1][0] = WMMA_BF16(a1, b0, acc[1][0]);
      acc[1][1] = WMMA_BF16(a1, b1, acc[1][1]);
    }
    __syncthreads();                     // all waves done reading buf (i&1)
    if (i + 2 < NT) issue_tile(Ag, Bg, As, Bs, i & 1, m0, n0, (i + 2) * 64);
  }
}

// ---------------- prep: fp32 -> bf16 (x straight, weights transposed) -------
__global__ __launch_bounds__(256) void prep_kernel(
    const float* __restrict__ x, const float* __restrict__ Wq,
    const float* __restrict__ Wk, const float* __restrict__ Wv,
    const float* __restrict__ Wo, bf16_t* __restrict__ xb,
    bf16_t* __restrict__ Wt) {
  long i = (long)blockIdx.x * 256 + threadIdx.x;
  if (i < (long)NTOK * DM) xb[i] = f2bf(x[i]);
  if (i < 4l * DM * DM) {
    int w = (int)(i >> 20);
    int o = (int)(i >> 10) & (DM - 1);
    int c = (int)i & (DM - 1);
    const float* W = (w == 0) ? Wq : (w == 1) ? Wk : (w == 2) ? Wv : Wo;
    Wt[i] = f2bf(W[(long)c * DM + o]);   // Wt[w][o][c] = W[c][o]
  }
}

// ---------------- fused QKV projection GEMM --------------------------------
__global__ __launch_bounds__(256) void qkv_gemm(
    const bf16_t* __restrict__ xb, const bf16_t* __restrict__ WtAll,
    const float* __restrict__ bq, const float* __restrict__ bk,
    const float* __restrict__ bv, bf16_t* __restrict__ Qb,
    bf16_t* __restrict__ Kb, bf16_t* __restrict__ Vtb) {
  __shared__ bf16_t As[2 * A_BUF];
  __shared__ bf16_t Bs[2 * B_BUF];
  const int tid = threadIdx.x;
  const int lane = tid & 31, wid = tid >> 5;
  const int wm = wid & 3, wn = wid >> 2;
  const int mode = blockIdx.z;
  const int m0 = blockIdx.x * 128, n0 = blockIdx.y * 64;
  const bf16_t* Wt = WtAll + (size_t)mode * DM * DM;
  const float* bias = (mode == 0) ? bq : (mode == 1) ? bk : bv;

  v8f acc[2][2] = {};
  gemm_core(xb, Wt, As, Bs, m0, n0, lane, wm, wn, acc);

  const int mloc = lane & 15, hf = lane >> 4;
#pragma unroll
  for (int mi = 0; mi < 2; ++mi)
#pragma unroll
    for (int ni = 0; ni < 2; ++ni)
#pragma unroll
      for (int r = 0; r < 8; ++r) {
        int gm = m0 + wm * 32 + mi * 16 + r + 8 * hf;   // token
        int gn = n0 + wn * 32 + ni * 16 + mloc;         // output feature
        float val = acc[mi][ni][r] + bias[gn];
        int b = gm >> 11, s = gm & (SQ - 1);
        int hh = gn >> 6, d = gn & (DK - 1);
        if (mode == 2)
          Vtb[((size_t)(b * NH + hh) * DK + d) * SQ + s] = f2bf(val);
        else {
          bf16_t* dst = (mode == 0) ? Qb : Kb;
          dst[((size_t)(b * NH + hh) * SQ + s) * DK + d] = f2bf(val);
        }
      }
}

// ---------------- flash attention ------------------------------------------
// grid: (SQ/128, B*H); 8 waves x 16 queries; kv tiles of 32 keys.
__global__ __launch_bounds__(256) void attn_kernel(
    const bf16_t* __restrict__ Qb, const bf16_t* __restrict__ Kb,
    const bf16_t* __restrict__ Vtb, bf16_t* __restrict__ Ob) {
  __shared__ bf16_t Plds[8][16 * 40];
  const int tid = threadIdx.x;
  const int lane = tid & 31, wid = tid >> 5;
  const int mloc = lane & 15, hf = lane >> 4;
  const int bh = blockIdx.y;
  const int qBase = blockIdx.x * 128 + wid * 16;
  const bf16_t* Qp = Qb + (size_t)bh * SQ * DK;
  const bf16_t* Kp = Kb + (size_t)bh * SQ * DK;
  const bf16_t* Vp = Vtb + (size_t)bh * DK * SQ;

  v16bf aq[2];
#pragma unroll
  for (int j = 0; j < 2; ++j) {
    const bf16_t* qr = Qp + (size_t)(qBase + mloc) * DK + 32 * j;
    AF f;
    f.h[0] = *(const v8bf*)(qr + 8 * hf);
    f.h[1] = *(const v8bf*)(qr + 16 + 8 * hf);
    aq[j] = f.v;
  }
  v8f acc[4] = {};
  float mr[8], lr[8];
#pragma unroll
  for (int r = 0; r < 8; ++r) { mr[r] = -3.0e38f; lr[r] = 0.f; }
  const float scale = 0.125f;   // 1/sqrt(64)
  bf16_t* P = &Plds[wid][0];

  for (int kv = 0; kv < SQ; kv += 32) {
    v8f sc[2] = {};
#pragma unroll
    for (int nt = 0; nt < 2; ++nt)
#pragma unroll
      for (int j = 0; j < 2; ++j) {
        const bf16_t* kr = Kp + (size_t)(kv + nt * 16 + mloc) * DK + 32 * j + 8 * hf;
        AF f;
        f.h[0] = *(const v8bf*)(kr);
        f.h[1] = *(const v8bf*)(kr + 16);
        sc[nt] = WMMA_BF16(aq[j], f.v, sc[nt]);
      }
    // online softmax: row M = r + 8*hf lives across the 16 lanes of this half
#pragma unroll
    for (int r = 0; r < 8; ++r) {
      float s0 = sc[0][r] * scale, s1 = sc[1][r] * scale;
      float t = fmaxf(s0, s1);
#pragma unroll
      for (int off = 1; off < 16; off <<= 1) t = fmaxf(t, __shfl_xor(t, off, 32));
      float mn = fmaxf(mr[r], t);
      float al = __expf(mr[r] - mn);
      float p0 = __expf(s0 - mn), p1 = __expf(s1 - mn);
      float ls = p0 + p1;
#pragma unroll
      for (int off = 1; off < 16; off <<= 1) ls += __shfl_xor(ls, off, 32);
      lr[r] = lr[r] * al + ls;
      mr[r] = mn;
#pragma unroll
      for (int t4 = 0; t4 < 4; ++t4) acc[t4][r] = acc[t4][r] * al;
      int row = r + 8 * hf;
      P[row * 40 + mloc] = f2bf(p0);
      P[row * 40 + 16 + mloc] = f2bf(p1);
    }
    // C-layout -> A-frag re-swizzle through per-wave LDS (same-wave DS in-order)
    AF fp;
    fp.h[0] = *(const v8bf*)(P + mloc * 40 + 8 * hf);
    fp.h[1] = *(const v8bf*)(P + mloc * 40 + 16 + 8 * hf);
#pragma unroll
    for (int t4 = 0; t4 < 4; ++t4) {
      const bf16_t* vr = Vp + (size_t)(t4 * 16 + mloc) * SQ + kv + 8 * hf;
      AF fv;
      fv.h[0] = *(const v8bf*)(vr);
      fv.h[1] = *(const v8bf*)(vr + 16);
      acc[t4] = WMMA_BF16(fp.v, fv.v, acc[t4]);
    }
  }
  const int b = bh >> 4, hh = bh & 15;
#pragma unroll
  for (int r = 0; r < 8; ++r) {
    float inv = 1.f / lr[r];
    int s = qBase + r + 8 * hf;
    size_t base = ((size_t)(b * SQ + s)) * DM + hh * DK;
#pragma unroll
    for (int t4 = 0; t4 < 4; ++t4)
      Ob[base + t4 * 16 + mloc] = f2bf(acc[t4][r] * inv);
  }
}

// ---------------- output projection (fp32 epilogue) ------------------------
__global__ __launch_bounds__(256) void out_gemm(
    const bf16_t* __restrict__ Ob, const bf16_t* __restrict__ Wto,
    const float* __restrict__ bo, float* __restrict__ out) {
  __shared__ bf16_t As[2 * A_BUF];
  __shared__ bf16_t Bs[2 * B_BUF];
  const int tid = threadIdx.x;
  const int lane = tid & 31, wid = tid >> 5;
  const int wm = wid & 3, wn = wid >> 2;
  const int m0 = blockIdx.x * 128, n0 = blockIdx.y * 64;

  v8f acc[2][2] = {};
  gemm_core(Ob, Wto, As, Bs, m0, n0, lane, wm, wn, acc);

  const int mloc = lane & 15, hf = lane >> 4;
#pragma unroll
  for (int mi = 0; mi < 2; ++mi)
#pragma unroll
    for (int ni = 0; ni < 2; ++ni)
#pragma unroll
      for (int r = 0; r < 8; ++r) {
        int gm = m0 + wm * 32 + mi * 16 + r + 8 * hf;
        int gn = n0 + wn * 32 + ni * 16 + mloc;
        out[(size_t)gm * DM + gn] = acc[mi][ni][r] + bo[gn];
      }
}

extern "C" void kernel_launch(void* const* d_in, const int* in_sizes, int n_in,
                              void* d_out, int out_size, void* d_ws, size_t ws_size,
                              hipStream_t stream) {
  (void)in_sizes; (void)n_in; (void)out_size; (void)ws_size;
  const float* x  = (const float*)d_in[0];
  const float* Wq = (const float*)d_in[1];
  const float* bq = (const float*)d_in[2];
  const float* Wk = (const float*)d_in[3];
  const float* bk = (const float*)d_in[4];
  const float* Wv = (const float*)d_in[5];
  const float* bv = (const float*)d_in[6];
  const float* Wo = (const float*)d_in[7];
  const float* bo = (const float*)d_in[8];

  // workspace layout (bf16): xb | Wt(q,k,v,o transposed) | Q | K | Vt | O  ~92 MB
  bf16_t* xb  = (bf16_t*)d_ws;
  bf16_t* Wt  = xb + (size_t)NTOK * DM;
  bf16_t* Qb  = Wt + (size_t)4 * DM * DM;
  bf16_t* Kb  = Qb + (size_t)NB * NH * SQ * DK;
  bf16_t* Vtb = Kb + (size_t)NB * NH * SQ * DK;
  bf16_t* Obf = Vtb + (size_t)NB * NH * SQ * DK;

  prep_kernel<<<(NTOK * DM) / 256, 256, 0, stream>>>(x, Wq, Wk, Wv, Wo, xb, Wt);
  qkv_gemm<<<dim3(NTOK / 128, DM / 64, 3), 256, 0, stream>>>(xb, Wt, bq, bk, bv, Qb, Kb, Vtb);
  attn_kernel<<<dim3(SQ / 128, NB * NH), 256, 0, stream>>>(Qb, Kb, Vtb, Obf);
  out_gemm<<<dim3(NTOK / 128, DM / 64), 256, 0, stream>>>(Obf, Wt + (size_t)3 * DM * DM, bo,
                                                          (float*)d_out);
}
